// Attention_67594195305169
// MI455X (gfx1250) — matrix operand order repaired and checked by reading, not compiled, not run on hardware
//
#include <hip/hip_runtime.h>

typedef __attribute__((ext_vector_type(16))) _Float16 v16h;
typedef __attribute__((ext_vector_type(8)))  _Float16 v8h;
typedef __attribute__((ext_vector_type(8)))  float    v8f;
typedef __attribute__((ext_vector_type(4)))  float    v4f;

#define NSEQ  2048
#define DHEAD 64
#define TK    64      // keys per iteration
#define NITER (NSEQ / TK)
#define KROW  72      // padded LDS row stride in halves (64 + 8)

// Operand halves, contiguous:
// elements 0..7 from p[0..7], elements 8..15 from p[8..15]
static __device__ __forceinline__ v16h ld16gap(const _Float16* p) {
  v8h lo = *(const v8h*)(p);
  v8h hh = *(const v8h*)(p + 8);
  v16h r;
#pragma unroll
  for (int i = 0; i < 8; ++i) { r[i] = lo[i]; r[i + 8] = hh[i]; }
  return r;
}

static __device__ __forceinline__ v8f wmma16(v16h a, v16h b, v8f c) {
  return __builtin_amdgcn_wmma_f32_16x16x32_f16(false, a, false, b, (short)0, c,
                                                false, false);
}

static __device__ __forceinline__ unsigned pack2h(float a, float b) {
  union { _Float16 h[2]; unsigned u; } x;
  x.h[0] = (_Float16)a; x.h[1] = (_Float16)b;
  return x.u;  // v_cvt_pk_f16_f32
}

// Online-softmax update for one query tile: st[4] holds S^T (C-layout, log2
// domain). Updates running max/sum, rewrites st as P, builds the two P^T
// B-operands (k2 chunks of 32) via one half-wave exchange per chunk.
// Returns the accumulator rescale factor.
static __device__ __forceinline__ float softmax_update(v8f st[4], float& mrow,
                                                       float& lrow, int hi,
                                                       v16h bop[2]) {
  float mx = st[0][0];
#pragma unroll
  for (int j = 0; j < 4; ++j)
#pragma unroll
    for (int r = 0; r < 8; ++r) mx = fmaxf(mx, st[j][r]);
  mx = fmaxf(mx, __shfl_xor(mx, 16, 32));

  const float mn = fmaxf(mrow, mx);
  const float al = __builtin_amdgcn_exp2f(mrow - mn);
  mrow = mn;

  float rs = 0.0f;
#pragma unroll
  for (int j = 0; j < 4; ++j)
#pragma unroll
    for (int r = 0; r < 8; ++r) {
      const float p = __builtin_amdgcn_exp2f(st[j][r] - mn);
      st[j][r] = p;
      rs += p;
    }
  rs += __shfl_xor(rs, 16, 32);
  lrow = lrow * al + rs;

#pragma unroll
  for (int c = 0; c < 2; ++c) {
    unsigned own0[4], own1[4];
#pragma unroll
    for (int vv = 0; vv < 4; ++vv) {
      own0[vv] = pack2h(st[2 * c][2 * vv],     st[2 * c][2 * vv + 1]);
      own1[vv] = pack2h(st[2 * c + 1][2 * vv], st[2 * c + 1][2 * vv + 1]);
    }
    union { v16h v; unsigned u[8]; } bb;
#pragma unroll
    for (int vv = 0; vv < 4; ++vv) {
      const unsigned send = hi ? own0[vv] : own1[vv];
      const unsigned recv = (unsigned)__shfl_xor((int)send, 16, 32);
      bb.u[vv]     = hi ? recv : own0[vv];
      bb.u[vv + 4] = hi ? own1[vv] : recv;
    }
    bop[c] = bb.v;
  }
  return al;
}

__global__ __launch_bounds__(128) void fa_fwd_kernel(
    const float* __restrict__ Q, const float* __restrict__ K,
    const float* __restrict__ V, float* __restrict__ O) {
  __shared__ __attribute__((aligned(16))) _Float16 Ksh[TK * KROW];     // K tile row-major f16
  __shared__ __attribute__((aligned(16))) _Float16 Vtsh[DHEAD * KROW]; // V tile transposed f16

  const int tid  = threadIdx.x;
  const int lane = tid & 31;
  const int ln   = lane & 15;
  const int hi   = lane >> 4;

  const int qt = blockIdx.x & 15;        // 16 query tiles of 128 rows
  const int bh = blockIdx.x >> 4;        // flattened (b,h)
  const size_t head = (size_t)bh * (NSEQ * DHEAD);
  const int qrowA = qt * 128 + (tid >> 5) * 16;  // wave's first query tile
  const int qrowB = qrowA + 64;                  // wave's second query tile

  // ---- Q in B-operand layout, scaled by log2(e)/sqrt(D) (softmax in exp2).
  v16h qA0, qA1, qB0, qB1;
  {
    const float sc = 0.125f * 1.44269504088896f;
    const float* qa = Q + head + (size_t)(qrowA + ln) * DHEAD + 16 * hi;
    const float* qb = Q + head + (size_t)(qrowB + ln) * DHEAD + 16 * hi;
#pragma unroll
    for (int e = 0; e < 16; ++e) {
      qA0[e] = (_Float16)(qa[e]      * sc);
      qA1[e] = (_Float16)(qa[e + 32] * sc);
      qB0[e] = (_Float16)(qb[e]      * sc);
      qB1[e] = (_Float16)(qb[e + 32] * sc);
    }
  }

  v8f accA[4] = {}, accB[4] = {};   // O^T accumulators per query tile
  float mA = -1.0e30f, lA = 0.0f;
  float mB = -1.0e30f, lB = 0.0f;

  for (int kt = 0; kt < NITER; ++kt) {
    __syncthreads();  // previous iteration's LDS reads complete

    // ---- Stage K tile (64x64) row-major f16 (coalesced)
    {
      const int row = tid >> 1, seg = (tid & 1) * 32;
      const float* kp = K + head + (size_t)(kt * TK + row) * DHEAD + seg;
      _Float16* dst = Ksh + row * KROW + seg;
#pragma unroll
      for (int i = 0; i < 32; ++i) dst[i] = (_Float16)kp[i];
      if (kt + 1 < NITER) __builtin_prefetch(kp + TK * DHEAD, 0, 3);
    }
    // ---- Stage V tile transposed: Vtsh[d][k2]
    {
      const int d = tid >> 1, seg = (tid & 1) * 32;
      const float* vp = V + head + (size_t)(kt * TK + seg) * DHEAD + d;
      _Float16* dst = Vtsh + d * KROW + seg;
#pragma unroll
      for (int i = 0; i < 32; ++i) dst[i] = (_Float16)vp[(size_t)i * DHEAD];
      if (kt + 1 < NITER) __builtin_prefetch(vp + TK * DHEAD, 0, 3);
    }
    __syncthreads();

    // ---- S^T = K * Q^T for both query tiles; A-operands shared.
    v8f st0[4], st1[4];
#pragma unroll
    for (int j = 0; j < 4; ++j) {
      const _Float16* ar = Ksh + (j * 16 + ln) * KROW + 16 * hi;
      const v16h a0 = ld16gap(ar);
      const v16h a1 = ld16gap(ar + 32);
      v8f s0 = {}, s1 = {};
      s0 = wmma16(a0, qA0, s0); s0 = wmma16(a1, qA1, s0);
      s1 = wmma16(a0, qB0, s1); s1 = wmma16(a1, qB1, s1);
      st0[j] = s0; st1[j] = s1;
    }

    // ---- Online softmax + P^T operand build per query tile.
    v16h bopA[2], bopB[2];
    const float alA = softmax_update(st0, mA, lA, hi, bopA);
#pragma unroll
    for (int t = 0; t < 4; ++t)
#pragma unroll
      for (int r = 0; r < 8; ++r) accA[t][r] *= alA;
    const float alB = softmax_update(st1, mB, lB, hi, bopB);
#pragma unroll
    for (int t = 0; t < 4; ++t)
#pragma unroll
      for (int r = 0; r < 8; ++r) accB[t][r] *= alB;

    // ---- O^T += V^T * P^T for both query tiles; A-operands shared.
#pragma unroll
    for (int t = 0; t < 4; ++t) {
      const _Float16* ar = Vtsh + (16 * t + ln) * KROW + 16 * hi;
      const v16h a0 = ld16gap(ar);
      const v16h a1 = ld16gap(ar + 32);
      v8f a = accA[t];
      a = wmma16(a0, bopA[0], a); a = wmma16(a1, bopA[1], a);
      accA[t] = a;
      v8f b = accB[t];
      b = wmma16(a0, bopB[0], b); b = wmma16(a1, bopB[1], b);
      accB[t] = b;
    }
  }

  // ---- Epilogue: lane (ln,hi) holds out row q, d = 16t + 8hi + r.
  {
    const float invl = 1.0f / lA;
    float* op = O + head + (size_t)(qrowA + ln) * DHEAD + 8 * hi;
#pragma unroll
    for (int t = 0; t < 4; ++t) {
      v4f lo = {accA[t][0] * invl, accA[t][1] * invl, accA[t][2] * invl, accA[t][3] * invl};
      v4f hh = {accA[t][4] * invl, accA[t][5] * invl, accA[t][6] * invl, accA[t][7] * invl};
      *(v4f*)(op + 16 * t)     = lo;
      *(v4f*)(op + 16 * t + 4) = hh;
    }
  }
  {
    const float invl = 1.0f / lB;
    float* op = O + head + (size_t)(qrowB + ln) * DHEAD + 8 * hi;
#pragma unroll
    for (int t = 0; t < 4; ++t) {
      v4f lo = {accB[t][0] * invl, accB[t][1] * invl, accB[t][2] * invl, accB[t][3] * invl};
      v4f hh = {accB[t][4] * invl, accB[t][5] * invl, accB[t][6] * invl, accB[t][7] * invl};
      *(v4f*)(op + 16 * t)     = lo;
      *(v4f*)(op + 16 * t + 4) = hh;
    }
  }
}

extern "C" void kernel_launch(void* const* d_in, const int* in_sizes, int n_in,
                              void* d_out, int out_size, void* d_ws, size_t ws_size,
                              hipStream_t stream) {
  (void)in_sizes; (void)n_in; (void)out_size; (void)d_ws; (void)ws_size;
  const float* q = (const float*)d_in[0];
  const float* k = (const float*)d_in[1];
  const float* v = (const float*)d_in[2];
  float* out = (float*)d_out;
  // grid: (B*H)=32 heads * 16 query tiles of 128 rows; block: 4 waves
  fa_fwd_kernel<<<dim3(512), dim3(128), 0, stream>>>(q, k, v, out);
}
